// Attention_57715770523708
// MI455X (gfx1250) — matrix operand order repaired and checked by reading, not compiled, not run on hardware
//
#include <hip/hip_runtime.h>

typedef __bf16 v16bf __attribute__((ext_vector_type(16)));
typedef __bf16 v8bf  __attribute__((ext_vector_type(8)));
typedef __bf16 v2bf  __attribute__((ext_vector_type(2)));
typedef float  v8f   __attribute__((ext_vector_type(8)));

constexpr int NB = 2;      // batch
constexpr int NH = 16;     // heads (= waves per workgroup)
constexpr int NS = 2048;   // sequence
constexpr int ND = 64;     // feature
constexpr int QT = 16;     // q rows per workgroup (one WMMA M tile)
constexpr int KT = 32;     // k columns per iteration (two WMMA N tiles)

// exp(s/8) == exp2(s * log2(e)/8)
constexpr float EXP_SCALE = 0.18033688011112042f;

static __device__ __forceinline__ v8f wmma_bf16(v16bf a, v16bf b, v8f c) {
    return __builtin_amdgcn_wmma_f32_16x16x32_bf16(false, a, false, b,
                                                   (short)0, c, false, false);
}

// ---- prepass 1: f32 -> bf16, same layout (for K) ----
__global__ __launch_bounds__(256)
void cvt_bf16_kernel(const float* __restrict__ src, __bf16* __restrict__ dst,
                     int npairs) {
    int i = blockIdx.x * 256 + threadIdx.x;
    if (i < npairs) {
        float2 f = ((const float2*)src)[i];
        v2bf o; o[0] = (__bf16)f.x; o[1] = (__bf16)f.y;
        ((v2bf*)dst)[i] = o;
    }
}

// ---- prepass 2: V [b,h,k,d] f32 -> Vt [b,h,d,k] bf16 (16x16 LDS tiles) ----
__global__ __launch_bounds__(256)
void transpose_v_kernel(const float* __restrict__ V, __bf16* __restrict__ Vt) {
    __shared__ __bf16 t[16][17];
    const int tid  = threadIdx.x;
    const int blk  = blockIdx.x;                 // NB*NH * (NS/16) * (ND/16)
    const int dt16 = blk & (ND / 16 - 1);        // 0..3
    const int kt16 = (blk >> 2) & (NS / 16 - 1); // 0..127
    const int bh   = blk >> 9;                   // 0..31

    const int dd = tid & 15, kk = tid >> 4;
    t[kk][dd] = (__bf16)V[((size_t)bh * NS + kt16 * 16 + kk) * ND + dt16 * 16 + dd];
    __syncthreads();
    const int kk2 = tid & 15, dd2 = tid >> 4;
    Vt[((size_t)bh * ND + dt16 * 16 + dd2) * NS + kt16 * 16 + kk2] = t[kk2][dd2];
}

// ---- main kernel: one workgroup = (b, 16-row q tile); wave i = head i ----
__global__ __launch_bounds__(512, 1)
void attn_headsoftmax_kernel(const float*  __restrict__ Q,
                             const __bf16* __restrict__ Kb,   // [B,H,S,D] bf16
                             const __bf16* __restrict__ Vt,   // [B,H,D,S] bf16
                             float* __restrict__ Out,         // [B,H,S,D]
                             float* __restrict__ Attn)        // [B,H,S,S]
{
    __shared__ float   red_sum[3][QT * KT];      // 6 KB, triple-buffered
    __shared__ __bf16  slab[NH][QT * KT];        // 16 KB: per-wave attn staging

    const int tid  = threadIdx.x;
    const int h    = tid >> 5;
    const int lane = tid & 31;
    const int b    = blockIdx.x / (NS / QT);
    const int qt   = blockIdx.x % (NS / QT);

    const size_t headoff = ((size_t)(b * NH + h)) * NS * ND;
    const float*  qh = Q  + headoff;
    const __bf16* kh = Kb + headoff;
    const __bf16* vh = Vt + headoff;             // [ND][NS]

    const int lrow = lane & 15;
    const int hi   = lane >> 4;
    const int kb0  = hi ? 8 : 0;

    // persistent Q A-fragments (A 16x32: lane L: M=L&15; e: K=hi*16+e)
    v16bf aQ0, aQ1;
    {
        const float* qrow = qh + (size_t)(qt * QT + lrow) * ND;
        #pragma unroll
        for (int e = 0; e < 16; ++e) {
            int kk = hi * 16 + e;
            aQ0[e] = (__bf16)qrow[kk];
            aQ1[e] = (__bf16)qrow[kk + 32];
        }
    }

    red_sum[0][tid] = 0.0f;
    red_sum[1][tid] = 0.0f;
    red_sum[2][tid] = 0.0f;
    __syncthreads();

    v8f acc[4] = {};
    float* attnQ = Attn + (((size_t)(b * NH + h)) * NS + (size_t)qt * QT) * NS;

    int cur = 0;
    for (int kt = 0; kt < NS; kt += KT) {
        // ---- scores: 16(q) x 32(k); B fragments are plain aligned bf16 loads ----
        v8f c0 = {}, c1 = {};
        #pragma unroll
        for (int ns = 0; ns < 2; ++ns) {
            const __bf16* krow = kh + (size_t)(kt + ns * 16 + lrow) * ND + hi * 16;
            v16bf bKlo = *(const v16bf*)(krow);        // d = hi*16 + e
            v16bf bKhi = *(const v16bf*)(krow + 32);   // d + 32
            v8f c = {};
            c = wmma_bf16(aQ0, bKlo, c);
            c = wmma_bf16(aQ1, bKhi, c);
            if (ns == 0) c0 = c; else c1 = c;
        }

        // ---- exp + cross-head sum (no max pass: |s| <= ~8, exp safe in f32) ----
        float e0[8], e1[8];
        #pragma unroll
        for (int r = 0; r < 8; ++r) {
            e0[r] = __builtin_amdgcn_exp2f(c0[r] * EXP_SCALE);
            e1[r] = __builtin_amdgcn_exp2f(c1[r] * EXP_SCALE);
            int q = hi ? (8 + r) : r;
            atomicAdd(&red_sum[cur][q * KT + lrow],      e0[r]);
            atomicAdd(&red_sum[cur][q * KT + 16 + lrow], e1[r]);
        }
        __syncthreads();   // single barrier per iteration (triple buffer)

        // reset buffer for iteration i+2 (its readers finished before this barrier)
        int nxt2 = cur + 2; if (nxt2 >= 3) nxt2 -= 3;
        red_sum[nxt2][tid] = 0.0f;

        // ---- normalize, stream attn to HBM (NT), stage bf16 tile to own slab ----
        #pragma unroll
        for (int r = 0; r < 8; ++r) {
            int q = hi ? (8 + r) : r;
            float i0 = __builtin_amdgcn_rcpf(red_sum[cur][q * KT + lrow]);
            float i1 = __builtin_amdgcn_rcpf(red_sum[cur][q * KT + 16 + lrow]);
            float a0 = e0[r] * i0;
            float a1 = e1[r] * i1;
            __builtin_nontemporal_store(a0, attnQ + (size_t)q * NS + kt + lrow);
            __builtin_nontemporal_store(a1, attnQ + (size_t)q * NS + kt + 16 + lrow);
            slab[h][q * KT + lrow]      = (__bf16)a0;
            slab[h][q * KT + 16 + lrow] = (__bf16)a1;
        }

        // ---- attn(16x32) @ V(32x64); A re-layout via own LDS slab ----
        v8bf alo = *(const v8bf*)&slab[h][lrow * KT + hi * 16];
        v8bf ahi = *(const v8bf*)&slab[h][lrow * KT + hi * 16 + 8];
        v16bf aA = __builtin_shufflevector(alo, ahi,
                                           0, 1, 2, 3, 4, 5, 6, 7,
                                           8, 9, 10, 11, 12, 13, 14, 15);
        #pragma unroll
        for (int d = 0; d < 4; ++d) {
            // B 32x16: lane: col = d*16+lrow; e: row k = hi*16+e  -> Vt row slice
            const __bf16* vrow = vh + (size_t)(d * 16 + lrow) * NS + kt + hi * 16;
            v16bf bV = *(const v16bf*)vrow;
            acc[d] = wmma_bf16(aA, bV, acc[d]);
        }

        cur = cur + 1; if (cur >= 3) cur -= 3;
    }

    // ---- write out: [B,H,S,D] ----
    float* outBase = Out + (((size_t)(b * NH + h)) * NS + (size_t)qt * QT) * ND;
    #pragma unroll
    for (int d = 0; d < 4; ++d) {
        #pragma unroll
        for (int r = 0; r < 8; ++r) {
            int q = hi ? (8 + r) : r;
            __builtin_nontemporal_store(acc[d][r], outBase + q * ND + d * 16 + lrow);
        }
    }
}

extern "C" void kernel_launch(void* const* d_in, const int* in_sizes, int n_in,
                              void* d_out, int out_size, void* d_ws, size_t ws_size,
                              hipStream_t stream) {
    const float* q = (const float*)d_in[0];
    const float* k = (const float*)d_in[1];
    const float* v = (const float*)d_in[2];
    // d_in[3] = feature_size (scalar, fixed 64) — baked into constants.

    float* out  = (float*)d_out;
    float* attn = out + (size_t)NB * NH * NS * ND;   // tuple: (out, attn) flat

    const size_t nelem = (size_t)NB * NH * NS * ND;  // 4.19M per tensor
    __bf16* kbf = (__bf16*)d_ws;                     // 8.4 MB
    __bf16* vtr = kbf + nelem;                       // 8.4 MB (transposed)

    // prepass: K -> bf16 (same layout); V -> bf16 transposed [b,h,d,k]
    {
        int npairs = (int)(nelem / 2);
        dim3 grid((npairs + 255) / 256), block(256);
        hipLaunchKernelGGL(cvt_bf16_kernel, grid, block, 0, stream, k, kbf, npairs);
    }
    {
        dim3 grid(NB * NH * (NS / 16) * (ND / 16)), block(256);   // 16384 tiles
        hipLaunchKernelGGL(transpose_v_kernel, grid, block, 0, stream, v, vtr);
    }

    // main: 256 workgroups x 16 wave32 (one wave per head)
    dim3 grid(NB * (NS / QT)), block(512);
    hipLaunchKernelGGL(attn_headsoftmax_kernel, grid, block, 0, stream,
                       q, kbf, vtr, out, attn);
}
